// MambaSpatial_29102698398069
// MI455X (gfx1250) — matrix operand order repaired and checked by reading, not compiled, not run on hardware
//
#include <hip/hip_runtime.h>
#include <hip/hip_bf16.h>
#include <math.h>

// ---------------------------------------------------------------------------
// MambaSpatial forward for MI455X (gfx1250, wave32).
// Dense projections use v_wmma_f32_16x16x32_f16 (f16 in / f32 accumulate);
// selective scan runs in f32, 16 lanes = N states, half-wave shuffle reduce.
// ---------------------------------------------------------------------------

typedef __attribute__((ext_vector_type(16))) _Float16 v16h;
typedef __attribute__((ext_vector_type(8)))  float    v8f;

#define DEV __device__ __forceinline__

constexpr int Bn = 32;            // batch
constexpr int Cn = 96;            // C_IN
constexpr int Hn = 48;
constexpr int Wn = 48;
constexpr int Dn = 96;            // D
constexpr int Kn = 2;             // directions
constexpr int Nn = 16;            // state size
constexpr int Rn = 6;             // dt rank
constexpr int Ln = Hn * Wn;       // 2304
constexpr int CD = Rn + 2 * Nn;   // 38
constexpr int NPAD = 4096;        // bitonic pad of Ln

// ---------------------------------------------------------------------------
// WMMA helpers (ISA 7.12.2 layouts, wave32)
// ---------------------------------------------------------------------------

// A fragment: 16x32 f16. Lane L holds row M=L&15. Lower lanes: K 0-7 & 16-23,
// upper lanes: K 8-15 & 24-31. Source is f32 row-major, converted in-register.
DEV v16h load_a_frag(const float* A, int lda, int lane) {
  const float* p = A + (size_t)(lane & 15) * lda + ((lane >> 4) << 3);
  v16h f;
#pragma unroll
  for (int i = 0; i < 8; ++i) f[i] = (_Float16)p[i];
#pragma unroll
  for (int i = 0; i < 8; ++i) f[i + 8] = (_Float16)p[i + 16];
  return f;
}

// B fragment: 32x16 f16, taken from a row-major [N,K] weight matrix (so K is
// contiguous per lane). Lane L holds col N=L&15; lower lanes K 0-7 & 16-23,
// upper lanes K 8-15 & 24-31. nmax masks invalid N rows (x_proj has N=38).
DEV v16h load_b_frag(const float* Bw, int ldb, int lane, int nmax) {
  int n = lane & 15;
  v16h f;
  if (n < nmax) {
    const float* p = Bw + (size_t)n * ldb + ((lane >> 4) << 3);
#pragma unroll
    for (int i = 0; i < 16; ++i) f[i] = (_Float16)p[i < 8 ? i : i + 8];
  } else {
#pragma unroll
    for (int i = 0; i < 16; ++i) f[i] = (_Float16)0.0f;
  }
  return f;
}

// One 16x16 output tile over K=96 (3 x v_wmma_f32_16x16x32_f16).
DEV v8f gemm_tile_k96(const float* Atile, int lda, const float* Btile, int ldb,
                      int lane, int nmax) {
  v8f acc = {0.f, 0.f, 0.f, 0.f, 0.f, 0.f, 0.f, 0.f};
#pragma unroll
  for (int kk = 0; kk < 96; kk += 32) {
    v16h a = load_a_frag(Atile + kk, lda, lane);
    v16h b = load_b_frag(Btile + kk, ldb, lane, nmax);
    acc = __builtin_amdgcn_wmma_f32_16x16x32_f16(
        /*neg_a=*/false, a, /*neg_b=*/false, b,
        /*c_mod=*/(short)0, acc, /*reuse_a=*/false, /*reuse_b=*/false);
  }
  return acc;
}

DEV float wave_reduce_add32(float v) {
#pragma unroll
  for (int m = 16; m >= 1; m >>= 1) v += __shfl_xor(v, m, 32);
  return v;
}

DEV float silu(float x) { return x / (1.f + __expf(-x)); }

// ---------------------------------------------------------------------------
// K0: zero small accumulators
// ---------------------------------------------------------------------------
__global__ void zero_kernel(float* p, int n) {
  int i = blockIdx.x * blockDim.x + threadIdx.x;
  if (i < n) p[i] = 0.f;
}

// ---------------------------------------------------------------------------
// K1: transpose x [b,c,l] -> xT [b,l,c] (GEMM A layout)
// ---------------------------------------------------------------------------
__global__ void transpose_kernel(const float* __restrict__ x, float* __restrict__ xT) {
  int t = blockIdx.x * blockDim.x + threadIdx.x;
  if (t >= Bn * Ln * Cn) return;
  int c = t % Cn;
  int l = (t / Cn) % Ln;
  int b = t / (Cn * Ln);
  xT[t] = x[((size_t)b * Cn + c) * Ln + l];
}

// ---------------------------------------------------------------------------
// K2: in_proj GEMM: x_feat[b,l,d] = xT[b,l,:] @ W[d,:]^T + bias[d]
// ---------------------------------------------------------------------------
__global__ void gemm_in_proj_kernel(const float* __restrict__ xT,
                                    const float* __restrict__ W,
                                    const float* __restrict__ bias,
                                    float* __restrict__ x_feat) {
  int wid = (blockIdx.x * blockDim.x + threadIdx.x) >> 5;
  int lane = threadIdx.x & 31;
  const int MT = (Bn * Ln) / 16, NT = Dn / 16;
  if (wid >= MT * NT) return;
  int tm = wid / NT, tn = wid % NT;
  v8f acc = gemm_tile_k96(xT + (size_t)tm * 16 * Cn, Cn, W + (size_t)tn * 16 * Cn, Cn, lane, 16);
  int n = tn * 16 + (lane & 15);
  float bv = bias[n];
  int mbase = tm * 16 + ((lane >> 4) << 3);
#pragma unroll
  for (int r = 0; r < 8; ++r)
    x_feat[(size_t)(mbase + r) * Dn + n] = acc[r] + bv;
}

// ---------------------------------------------------------------------------
// K3: depthwise 3x3 conv + SiLU: x_feat[b,l,d] -> x_conv[b,d,l]
// ---------------------------------------------------------------------------
__global__ void dwconv_silu_kernel(const float* __restrict__ x_feat,
                                   const float* __restrict__ w,
                                   const float* __restrict__ bias,
                                   float* __restrict__ x_conv) {
  int t = blockIdx.x * blockDim.x + threadIdx.x;
  if (t >= Bn * Dn * Ln) return;
  int l = t % Ln;
  int d = (t / Ln) % Dn;
  int b = t / (Ln * Dn);
  int hh = l / Wn, ww = l % Wn;
  float acc = bias[d];
#pragma unroll
  for (int dh = -1; dh <= 1; ++dh) {
#pragma unroll
    for (int dw = -1; dw <= 1; ++dw) {
      int h2 = hh + dh, w2 = ww + dw;
      if (h2 >= 0 && h2 < Hn && w2 >= 0 && w2 < Wn)
        acc += x_feat[((size_t)b * Ln + h2 * Wn + w2) * Dn + d] *
               w[d * 9 + (dh + 1) * 3 + (dw + 1)];
    }
  }
  x_conv[t] = silu(acc);   // t == (b*Dn+d)*Ln + l
}

// ---------------------------------------------------------------------------
// K4: cosine similarity vs center pixel
// ---------------------------------------------------------------------------
__global__ void sim_kernel(const float* __restrict__ x_conv, float* __restrict__ sim) {
  __shared__ float cf[Dn];
  __shared__ float cfn;
  int b = blockIdx.y;
  const float* base = x_conv + (size_t)b * Dn * Ln;
  const int lc = (Hn / 2) * Wn + (Wn / 2);
  if (threadIdx.x < Dn) cf[threadIdx.x] = base[(size_t)threadIdx.x * Ln + lc];
  __syncthreads();
  if (threadIdx.x == 0) {
    float s = 0.f;
    for (int d = 0; d < Dn; ++d) s += cf[d] * cf[d];
    cfn = fmaxf(sqrtf(s), 1e-12f);
  }
  __syncthreads();
  int l = blockIdx.x * blockDim.x + threadIdx.x;
  if (l >= Ln) return;
  float dot = 0.f, sq = 0.f;
  for (int d = 0; d < Dn; ++d) {
    float v = base[(size_t)d * Ln + l];
    dot += v * cf[d];
    sq += v * v;
  }
  sim[b * Ln + l] = dot / (fmaxf(sqrtf(sq), 1e-12f) * cfn);
}

// ---------------------------------------------------------------------------
// K5: per-batch bitonic argsort, descending sim, stable on ties (idx asc)
// ---------------------------------------------------------------------------
DEV bool sort_before(float ka, int ia, float kb, int ib) {
  return (ka > kb) || (ka == kb && ia < ib);
}

__global__ void __launch_bounds__(1024) sort_kernel(const float* __restrict__ sim,
                                                    int* __restrict__ sidx) {
  __shared__ float k[NPAD];
  __shared__ int v[NPAD];
  int b = blockIdx.x;
  for (int i = threadIdx.x; i < NPAD; i += blockDim.x) {
    k[i] = (i < Ln) ? sim[b * Ln + i] : -2.0f;   // sims are in [-1,1]
    v[i] = i;
  }
  for (int size = 2; size <= NPAD; size <<= 1) {
    for (int stride = size >> 1; stride > 0; stride >>= 1) {
      __syncthreads();
      for (int t = threadIdx.x; t < NPAD / 2; t += blockDim.x) {
        int pos = 2 * t - (t & (stride - 1));
        int i = pos, j = pos + stride;
        bool fwd = ((pos & size) == 0);
        float ki = k[i], kj = k[j];
        int vi = v[i], vj = v[j];
        bool sw = fwd ? sort_before(kj, vj, ki, vi) : sort_before(ki, vi, kj, vj);
        if (sw) { k[i] = kj; k[j] = ki; v[i] = vj; v[j] = vi; }
      }
    }
  }
  __syncthreads();
  for (int i = threadIdx.x; i < Ln; i += blockDim.x) sidx[b * Ln + i] = v[i];
}

// ---------------------------------------------------------------------------
// K6: gather sorted sequences: xs_t[b,k,l,d]  (k=1 is the flipped order)
// ---------------------------------------------------------------------------
__global__ void gather_xs_kernel(const float* __restrict__ x_conv,
                                 const int* __restrict__ sidx,
                                 float* __restrict__ xs_t) {
  int t = blockIdx.x * blockDim.x + threadIdx.x;
  if (t >= Bn * Kn * Ln * Dn) return;
  int d = t % Dn;
  int l = (t / Dn) % Ln;
  int kk = (t / (Dn * Ln)) % Kn;
  int b = t / (Dn * Ln * Kn);
  int ls = (kk == 0) ? l : (Ln - 1 - l);
  int j = sidx[b * Ln + ls];
  xs_t[t] = x_conv[((size_t)b * Dn + d) * Ln + j];
}

// ---------------------------------------------------------------------------
// K7: x_proj GEMM: xdbl[b,k,l,c] = xs_t[b,k,l,:] @ x_proj_w[k][c,:]^T  (c<38)
// ---------------------------------------------------------------------------
__global__ void gemm_xdbl_kernel(const float* __restrict__ xs_t,
                                 const float* __restrict__ xpw,
                                 float* __restrict__ xdbl) {
  int wid = (blockIdx.x * blockDim.x + threadIdx.x) >> 5;
  int lane = threadIdx.x & 31;
  const int MT = (Bn * Kn * Ln) / 16, NT = 3;   // N padded 38 -> 48
  if (wid >= MT * NT) return;
  int tm = wid / NT, tn = wid % NT;
  int row0 = tm * 16;
  int kk = (row0 / Ln) & (Kn - 1);
  int nmax = CD - tn * 16; nmax = nmax > 16 ? 16 : nmax;
  v8f acc = gemm_tile_k96(xs_t + (size_t)row0 * Dn, Dn,
                          xpw + (size_t)kk * CD * Dn + (size_t)tn * 16 * Dn, Dn,
                          lane, nmax);
  int n = tn * 16 + (lane & 15);
  if (n >= CD) return;
  int mbase = row0 + ((lane >> 4) << 3);
#pragma unroll
  for (int r = 0; r < 8; ++r)
    xdbl[(size_t)(mbase + r) * CD + n] = acc[r];
}

// ---------------------------------------------------------------------------
// K8: dt projection + softplus: dts_t[b,k,l,d]
// ---------------------------------------------------------------------------
__global__ void dtproj_kernel(const float* __restrict__ xdbl,
                              const float* __restrict__ dtw,
                              const float* __restrict__ dtb,
                              float* __restrict__ dts_t) {
  int t = blockIdx.x * blockDim.x + threadIdx.x;
  if (t >= Bn * Kn * Ln * Dn) return;
  int d = t % Dn;
  int l = (t / Dn) % Ln;
  int kk = (t / (Dn * Ln)) % Kn;
  int b = t / (Dn * Ln * Kn);
  const float* row = xdbl + ((size_t)((b * Kn + kk) * Ln) + l) * CD;
  const float* wr = dtw + ((size_t)kk * Dn + d) * Rn;
  float acc = dtb[kk * Dn + d];
#pragma unroll
  for (int r = 0; r < Rn; ++r) acc += row[r] * wr[r];
  dts_t[t] = (acc > 20.f) ? acc : log1pf(__expf(acc));
}

// ---------------------------------------------------------------------------
// K9: selective scan. Block = 256 threads = 8 waves; wave = 2 d-rows x 16 n.
// Grid = Bn*Kn*(Dn/16).
// ---------------------------------------------------------------------------
__global__ void scan_kernel(const float* __restrict__ dts_t,
                            const float* __restrict__ xdbl,
                            const float* __restrict__ xs_t,
                            const float* __restrict__ A_logs,
                            const float* __restrict__ Ds,
                            float* __restrict__ ys_t) {
  int blk = blockIdx.x;
  int dchunk = blk % (Dn / 16);
  int kk = (blk / (Dn / 16)) % Kn;
  int b = blk / ((Dn / 16) * Kn);
  int lane = threadIdx.x & 31;
  int wave = threadIdx.x >> 5;
  int d = dchunk * 16 + (wave << 1) + (lane >> 4);
  int n = lane & 15;

  float Aval = -__expf(A_logs[((size_t)(kk * Dn + d)) * Nn + n]);
  float Dval = Ds[kk * Dn + d];

  size_t rowbase = (size_t)(b * Kn + kk) * Ln;
  const float* dt_p = dts_t + rowbase * Dn + d;
  const float* x_p  = xs_t  + rowbase * Dn + d;
  const float* B_p  = xdbl  + rowbase * CD + Rn + n;
  const float* C_p  = B_p + Nn;
  float* y_p = ys_t + rowbase * Dn + d;

  float h = 0.f;
  for (int l = 0; l < Ln; ++l) {
    float dt = dt_p[(size_t)l * Dn];
    float xv = x_p[(size_t)l * Dn];
    float Bv = B_p[(size_t)l * CD];
    float Cv = C_p[(size_t)l * CD];
    float dA = __expf(dt * Aval);
    h = fmaf(dA, h, dt * xv * Bv);
    float yc = h * Cv;
    // reduce over the 16-lane n-group (xor masks < 16 stay within the half)
    yc += __shfl_xor(yc, 8, 32);
    yc += __shfl_xor(yc, 4, 32);
    yc += __shfl_xor(yc, 2, 32);
    yc += __shfl_xor(yc, 1, 32);
    if (n == 0) y_p[(size_t)l * Dn] = fmaf(Dval, xv, yc);
  }
}

// ---------------------------------------------------------------------------
// K10: combine directions, LayerNorm over d, scatter back to spatial order.
// One wave per sorted position l.  ycomb layout: [b, l', d]
// ---------------------------------------------------------------------------
__global__ void combine_ln_kernel(const float* __restrict__ ys_t,
                                  const int* __restrict__ sidx,
                                  const float* __restrict__ gamma,
                                  const float* __restrict__ beta,
                                  float* __restrict__ ycomb) {
  int gw = blockIdx.x * (blockDim.x >> 5) + (threadIdx.x >> 5);
  int lane = threadIdx.x & 31;
  int l = gw % Ln;
  int b = gw / Ln;
  if (b >= Bn) return;
  const float* y0 = ys_t + ((size_t)(b * Kn + 0) * Ln + l) * Dn;
  const float* y1 = ys_t + ((size_t)(b * Kn + 1) * Ln + (Ln - 1 - l)) * Dn;
  float v0 = y0[lane]      + y1[lane];
  float v1 = y0[lane + 32] + y1[lane + 32];
  float v2 = y0[lane + 64] + y1[lane + 64];
  float s  = wave_reduce_add32(v0 + v1 + v2);
  float sq = wave_reduce_add32(v0 * v0 + v1 * v1 + v2 * v2);
  float mu = s / (float)Dn;
  float inv = rsqrtf(sq / (float)Dn - mu * mu + 1e-5f);
  int j = sidx[b * Ln + l];
  float* dst = ycomb + ((size_t)b * Ln + j) * Dn;
  dst[lane]      = (v0 - mu) * inv * gamma[lane]      + beta[lane];
  dst[lane + 32] = (v1 - mu) * inv * gamma[lane + 32] + beta[lane + 32];
  dst[lane + 64] = (v2 - mu) * inv * gamma[lane + 64] + beta[lane + 64];
}

// ---------------------------------------------------------------------------
// K11: local depthwise conv + BN partial stats. Block per (b,d).
// ---------------------------------------------------------------------------
__global__ void localconv_kernel(const float* __restrict__ x_conv,
                                 const float* __restrict__ w,
                                 const float* __restrict__ bias,
                                 float* __restrict__ lc,
                                 float* __restrict__ bn_sum,
                                 float* __restrict__ bn_ssq) {
  int bd = blockIdx.x;
  int d = bd % Dn;
  const float* src = x_conv + (size_t)bd * Ln;
  float wl[9];
#pragma unroll
  for (int i = 0; i < 9; ++i) wl[i] = w[d * 9 + i];
  float bi = bias[d];
  float psum = 0.f, pssq = 0.f;
  for (int l = threadIdx.x; l < Ln; l += blockDim.x) {
    int hh = l / Wn, ww = l % Wn;
    float acc = bi;
#pragma unroll
    for (int dh = -1; dh <= 1; ++dh) {
#pragma unroll
      for (int dw = -1; dw <= 1; ++dw) {
        int h2 = hh + dh, w2 = ww + dw;
        if (h2 >= 0 && h2 < Hn && w2 >= 0 && w2 < Wn)
          acc += src[h2 * Wn + w2] * wl[(dh + 1) * 3 + (dw + 1)];
      }
    }
    lc[(size_t)bd * Ln + l] = acc;
    psum += acc;
    pssq += acc * acc;
  }
  psum = wave_reduce_add32(psum);
  pssq = wave_reduce_add32(pssq);
  __shared__ float ssum[8], sssq[8];
  int wave = threadIdx.x >> 5;
  if ((threadIdx.x & 31) == 0) { ssum[wave] = psum; sssq[wave] = pssq; }
  __syncthreads();
  if (threadIdx.x == 0) {
    float a = 0.f, c = 0.f;
    for (int i = 0; i < 8; ++i) { a += ssum[i]; c += sssq[i]; }
    atomicAdd(&bn_sum[d], a);
    atomicAdd(&bn_ssq[d], c);
  }
}

// ---------------------------------------------------------------------------
// K12: BN + SiLU + add mamba branch -> premul[b,l,d]
// ---------------------------------------------------------------------------
__global__ void bn_silu_add_kernel(const float* __restrict__ ycomb,
                                   const float* __restrict__ lc,
                                   const float* __restrict__ bn_sum,
                                   const float* __restrict__ bn_ssq,
                                   const float* __restrict__ bn_gamma,
                                   const float* __restrict__ bn_beta,
                                   float* __restrict__ premul) {
  int t = blockIdx.x * blockDim.x + threadIdx.x;
  if (t >= Bn * Ln * Dn) return;
  int d = t % Dn;
  int l = (t / Dn) % Ln;
  int b = t / (Dn * Ln);
  const float inv_cnt = 1.f / (float)(Bn * Ln);
  float mean = bn_sum[d] * inv_cnt;
  float var = bn_ssq[d] * inv_cnt - mean * mean;
  float scale = bn_gamma[d] * rsqrtf(var + 1e-5f);
  float shift = bn_beta[d] - mean * scale;
  float v = lc[((size_t)b * Dn + d) * Ln + l] * scale + shift;
  premul[t] = ycomb[t] + silu(v);
}

// ---------------------------------------------------------------------------
// K13: out_proj GEMM with transposed store: out[b,c,l]
// (skip-branch softmax over a singleton axis == 1.0, so no multiply needed)
// ---------------------------------------------------------------------------
__global__ void gemm_out_proj_kernel(const float* __restrict__ premul,
                                     const float* __restrict__ W,
                                     const float* __restrict__ bias,
                                     float* __restrict__ out) {
  int wid = (blockIdx.x * blockDim.x + threadIdx.x) >> 5;
  int lane = threadIdx.x & 31;
  const int MT = (Bn * Ln) / 16, NT = Cn / 16;
  if (wid >= MT * NT) return;
  int tm = wid / NT, tn = wid % NT;
  v8f acc = gemm_tile_k96(premul + (size_t)tm * 16 * Dn, Dn,
                          W + (size_t)tn * 16 * Dn, Dn, lane, 16);
  int n = tn * 16 + (lane & 15);
  float bv = bias[n];
  int mbase = tm * 16 + ((lane >> 4) << 3);
#pragma unroll
  for (int r = 0; r < 8; ++r) {
    int row = mbase + r;
    int b = row / Ln, l = row % Ln;
    out[((size_t)b * Cn + n) * Ln + l] = acc[r] + bv;
  }
}

// ---------------------------------------------------------------------------
// Host launcher
// ---------------------------------------------------------------------------
extern "C" void kernel_launch(void* const* d_in, const int* in_sizes, int n_in,
                              void* d_out, int out_size, void* d_ws, size_t ws_size,
                              hipStream_t stream) {
  (void)in_sizes; (void)n_in; (void)out_size; (void)ws_size;
  const float* x         = (const float*)d_in[0];
  const float* in_proj_w = (const float*)d_in[1];
  const float* in_proj_b = (const float*)d_in[2];
  // d_in[3] skip_w, d_in[4] skip_b: softmax over singleton axis => factor 1.0
  const float* conv2d_w  = (const float*)d_in[5];
  const float* conv2d_b  = (const float*)d_in[6];
  const float* local_w   = (const float*)d_in[7];
  const float* local_b   = (const float*)d_in[8];
  const float* bn_gamma  = (const float*)d_in[9];
  const float* bn_beta   = (const float*)d_in[10];
  const float* x_proj_w  = (const float*)d_in[11];
  const float* dt_projs_w= (const float*)d_in[12];
  const float* dt_projs_b= (const float*)d_in[13];
  const float* A_logs    = (const float*)d_in[14];
  const float* Ds        = (const float*)d_in[15];
  const float* norm_gamma= (const float*)d_in[16];
  const float* norm_beta = (const float*)d_in[17];
  const float* out_proj_w= (const float*)d_in[18];
  const float* out_proj_b= (const float*)d_in[19];
  float* out = (float*)d_out;

  // Workspace layout (byte offsets, overlaid by live range):
  const size_t SZ_BLD  = (size_t)Bn * Ln * Dn * 4;          // 28.3 MB
  const size_t SZ_BKLD = 2 * SZ_BLD;                        // 56.6 MB
  const size_t SZ_XDBL = (size_t)Bn * Kn * Ln * CD * 4;     // 22.4 MB
  char* ws = (char*)d_ws;
  float* xT     = (float*)(ws);                       // dead after K2
  float* ycomb  = xT;                                 // reborn at K10
  float* x_feat = (float*)(ws + SZ_BLD);              // dead after K3
  float* lc     = x_feat;                             // reborn at K11
  float* x_conv = (float*)(ws + 2 * SZ_BLD);
  float* xs_t   = (float*)(ws + 3 * SZ_BLD);          // dead after K9
  float* premul = xs_t;                               // reborn at K12
  float* xdbl   = (float*)(ws + 3 * SZ_BLD + SZ_BKLD);
  float* dts_t  = (float*)(ws + 3 * SZ_BLD + SZ_BKLD + SZ_XDBL);
  float* ys_t   = (float*)(ws + 3 * SZ_BLD + 2 * SZ_BKLD + SZ_XDBL);
  float* sim    = (float*)(ws + 3 * SZ_BLD + 3 * SZ_BKLD + SZ_XDBL);
  int*   sidx   = (int*)((char*)sim + (size_t)Bn * Ln * 4);
  float* bn_sum = (float*)((char*)sidx + (size_t)Bn * Ln * 4);
  float* bn_ssq = bn_sum + Dn;

  const int TPB = 256;
  // K0: BN accumulators must be zero every call (graph replay safe)
  zero_kernel<<<1, TPB, 0, stream>>>(bn_sum, 2 * Dn);
  // K1
  transpose_kernel<<<(Bn * Ln * Cn) / TPB, TPB, 0, stream>>>(x, xT);
  // K2 (4 waves/block, one 16x16 tile per wave)
  {
    int waves = ((Bn * Ln) / 16) * (Dn / 16);
    gemm_in_proj_kernel<<<(waves * 32 + 127) / 128, 128, 0, stream>>>(
        xT, in_proj_w, in_proj_b, x_feat);
  }
  // K3
  dwconv_silu_kernel<<<(Bn * Dn * Ln) / TPB, TPB, 0, stream>>>(
      x_feat, conv2d_w, conv2d_b, x_conv);
  // K4
  sim_kernel<<<dim3((Ln + TPB - 1) / TPB, Bn), TPB, 0, stream>>>(x_conv, sim);
  // K5
  sort_kernel<<<Bn, 1024, 0, stream>>>(sim, sidx);
  // K6
  gather_xs_kernel<<<(Bn * Kn * Ln * Dn) / TPB, TPB, 0, stream>>>(x_conv, sidx, xs_t);
  // K7
  {
    int waves = ((Bn * Kn * Ln) / 16) * 3;
    gemm_xdbl_kernel<<<(waves * 32 + 127) / 128, 128, 0, stream>>>(xs_t, x_proj_w, xdbl);
  }
  // K8
  dtproj_kernel<<<(Bn * Kn * Ln * Dn) / TPB, TPB, 0, stream>>>(
      xdbl, dt_projs_w, dt_projs_b, dts_t);
  // K9
  scan_kernel<<<Bn * Kn * (Dn / 16), 256, 0, stream>>>(
      dts_t, xdbl, xs_t, A_logs, Ds, ys_t);
  // K10 (8 waves/block, one l per wave)
  combine_ln_kernel<<<(Bn * Ln) / 8, 256, 0, stream>>>(
      ys_t, sidx, norm_gamma, norm_beta, ycomb);
  // K11
  localconv_kernel<<<Bn * Dn, 256, 0, stream>>>(
      x_conv, local_w, local_b, lc, bn_sum, bn_ssq);
  // K12
  bn_silu_add_kernel<<<(Bn * Ln * Dn) / TPB, TPB, 0, stream>>>(
      ycomb, lc, bn_sum, bn_ssq, bn_gamma, bn_beta, premul);
  // K13
  {
    int waves = ((Bn * Ln) / 16) * (Cn / 16);
    gemm_out_proj_kernel<<<(waves * 32 + 127) / 128, 128, 0, stream>>>(
        premul, out_proj_w, out_proj_b, out);
  }
}
